// LocalAggregator_52656299049415
// MI455X (gfx1250) — matrix operand order repaired and checked by reading, not compiled, not run on hardware
//
#include <hip/hip_runtime.h>

#define D_DIM 128
#define N_DIM 512
#define B_DIM 64
#define L_DIM 50
#define LEAKY 0.2f
#define MASKV -9000000000000000.0f

typedef __attribute__((ext_vector_type(16))) __bf16 v16bf;
typedef __attribute__((ext_vector_type(8)))  __bf16 v8bf;
typedef __attribute__((ext_vector_type(8)))  float  v8f;
typedef __attribute__((ext_vector_type(4)))  float  v4f;

// ---------------- Kernel 1: h_ds = mean(concat) @ wd  (B x D) ----------------
__global__ void k_hds(const float* __restrict__ hl, const float* __restrict__ htm,
                      const float* __restrict__ ie, const float* __restrict__ wd,
                      float* __restrict__ hds) {
    int b = blockIdx.x, d = threadIdx.x;
    __shared__ float mv[D_DIM];
    const float* p0 = hl  + (size_t)b * L_DIM * D_DIM + d;
    const float* p1 = htm + (size_t)b * L_DIM * D_DIM + d;
    const float* p2 = ie  + (size_t)b * L_DIM * D_DIM + d;
    float s = 0.f;
    for (int l = 0; l < L_DIM; ++l)
        s += p0[l * D_DIM] + p1[l * D_DIM] + p2[l * D_DIM];
    mv[d] = s * (1.0f / 150.0f);
    __syncthreads();
    float o = 0.f;
    for (int i = 0; i < D_DIM; ++i) o += mv[i] * wd[i * D_DIM + d];
    hds[b * D_DIM + d] = o;
}

// ------- Kernel 2: s_k[b,i] = sum_d h*h_ds*a_k ; hB = bf16(h) ; hT = bf16(h)^T -------
__global__ void k_prep(const float* __restrict__ h, const float* __restrict__ hds,
                       const float* __restrict__ a0, const float* __restrict__ a1,
                       const float* __restrict__ a2, const float* __restrict__ a3,
                       __bf16* __restrict__ hB, __bf16* __restrict__ hT,
                       float* __restrict__ sOut) {
    int t = blockIdx.x * blockDim.x + threadIdx.x;   // flat row id (b*N + i)
    int b = t >> 9, i = t & (N_DIM - 1);
    const float* hr = h + (size_t)t * D_DIM;
    const float* ds = hds + b * D_DIM;
    float s0 = 0, s1 = 0, s2 = 0, s3 = 0;
    for (int d = 0; d < D_DIM; ++d) {
        float hv = hr[d];
        __bf16 hb = (__bf16)hv;
        hB[(size_t)t * D_DIM + d] = hb;
        hT[((size_t)b * D_DIM + d) * N_DIM + i] = hb;
        float w = hv * ds[d];
        s0 += w * a0[d]; s1 += w * a1[d]; s2 += w * a2[d]; s3 += w * a3[d];
    }
    v4f sv = {s0, s1, s2, s3};
    *(v4f*)(sOut + (size_t)t * 4) = sv;
}

// ---------------- Kernel 3: fused scores -> select/mask -> softmax -> alpha@h ----------------
__global__ void __launch_bounds__(32)
k_attn(const float* __restrict__ h,
       const float* __restrict__ a0, const float* __restrict__ a1,
       const float* __restrict__ a2, const float* __restrict__ a3,
       const int*   __restrict__ adj,
       const __bf16* __restrict__ hB, const __bf16* __restrict__ hT,
       const float* __restrict__ sIn,
       float* __restrict__ out) {
    const int lane = threadIdx.x;
    const int m  = lane & 15;      // matrix row (A) / column (B,C,D)
    const int hf = lane >> 4;      // which half of the wave
    const int b  = blockIdx.x >> 5;
    const int i0 = (blockIdx.x & 31) << 4;

    __shared__ v16bf  uAp[4][4][32];       // per-lane A fragments of u_k = h*a_k (16KB)
    __shared__ float  es[16 * N_DIM];      // selected/masked scores (32KB)
    __shared__ __bf16 alph[16 * N_DIM];    // softmax result, bf16 (16KB)

    // ---- build A fragments: element e of lane covers K = kk*32 + hf*16 + e
    const float* hrow = h + (((size_t)b * N_DIM) + i0 + m) * D_DIM;
    #pragma unroll
    for (int k = 0; k < 4; ++k) {
        const float* ak = (k == 0) ? a0 : (k == 1) ? a1 : (k == 2) ? a2 : a3;
        #pragma unroll
        for (int kk = 0; kk < 4; ++kk) {
            v16bf v;
            #pragma unroll
            for (int e = 0; e < 16; ++e) {
                int d = kk * 32 + hf * 16 + e;
                v[e] = (__bf16)(hrow[d] * ak[d]);
            }
            uAp[k][kk][lane] = v;
        }
    }

    // per-lane loop-invariant s_i for the 8 rows this lane's C elements live in
    v4f si8[8];
    #pragma unroll
    for (int r = 0; r < 8; ++r)
        si8[r] = *(const v4f*)(sIn + (((size_t)b * N_DIM) + i0 + hf * 8 + r) * 4);
    __syncthreads();

    // ---- cross scores: 4 heads x (16 x 512) with K=128, fused epilogue per 16x16 j-tile
    for (int jt = 0; jt < 32; ++jt) {
        const int j0 = jt * 16;

        // hoist all VMEM for this tile: adj column, s_j, and the 4 B fragments
        const int* ap = adj + (size_t)b * N_DIM * N_DIM + (size_t)(i0 + hf * 8) * N_DIM + j0 + m;
        int adv[8];
        #pragma unroll
        for (int r = 0; r < 8; ++r) adv[r] = ap[(size_t)r * N_DIM];
        v4f sj = *(const v4f*)(sIn + (((size_t)b * N_DIM) + j0 + m) * 4);

        const __bf16* bbase = hB + (((size_t)b * N_DIM) + j0 + m) * D_DIM + hf * 16;
        v16bf bf0 = *(const v16bf*)(bbase);
        v16bf bf1 = *(const v16bf*)(bbase + 32);
        v16bf bf2 = *(const v16bf*)(bbase + 64);
        v16bf bf3 = *(const v16bf*)(bbase + 96);

        v8f c0{}, c1{}, c2{}, c3{};
        c0 = __builtin_amdgcn_wmma_f32_16x16x32_bf16(false, uAp[0][0][lane], false, bf0, (short)0, c0, false, false);
        c1 = __builtin_amdgcn_wmma_f32_16x16x32_bf16(false, uAp[1][0][lane], false, bf0, (short)0, c1, false, false);
        c2 = __builtin_amdgcn_wmma_f32_16x16x32_bf16(false, uAp[2][0][lane], false, bf0, (short)0, c2, false, false);
        c3 = __builtin_amdgcn_wmma_f32_16x16x32_bf16(false, uAp[3][0][lane], false, bf0, (short)0, c3, false, false);
        c0 = __builtin_amdgcn_wmma_f32_16x16x32_bf16(false, uAp[0][1][lane], false, bf1, (short)0, c0, false, false);
        c1 = __builtin_amdgcn_wmma_f32_16x16x32_bf16(false, uAp[1][1][lane], false, bf1, (short)0, c1, false, false);
        c2 = __builtin_amdgcn_wmma_f32_16x16x32_bf16(false, uAp[2][1][lane], false, bf1, (short)0, c2, false, false);
        c3 = __builtin_amdgcn_wmma_f32_16x16x32_bf16(false, uAp[3][1][lane], false, bf1, (short)0, c3, false, false);
        c0 = __builtin_amdgcn_wmma_f32_16x16x32_bf16(false, uAp[0][2][lane], false, bf2, (short)0, c0, false, false);
        c1 = __builtin_amdgcn_wmma_f32_16x16x32_bf16(false, uAp[1][2][lane], false, bf2, (short)0, c1, false, false);
        c2 = __builtin_amdgcn_wmma_f32_16x16x32_bf16(false, uAp[2][2][lane], false, bf2, (short)0, c2, false, false);
        c3 = __builtin_amdgcn_wmma_f32_16x16x32_bf16(false, uAp[3][2][lane], false, bf2, (short)0, c3, false, false);
        c0 = __builtin_amdgcn_wmma_f32_16x16x32_bf16(false, uAp[0][3][lane], false, bf3, (short)0, c0, false, false);
        c1 = __builtin_amdgcn_wmma_f32_16x16x32_bf16(false, uAp[1][3][lane], false, bf3, (short)0, c1, false, false);
        c2 = __builtin_amdgcn_wmma_f32_16x16x32_bf16(false, uAp[2][3][lane], false, bf3, (short)0, c2, false, false);
        c3 = __builtin_amdgcn_wmma_f32_16x16x32_bf16(false, uAp[3][3][lane], false, bf3, (short)0, c3, false, false);

        // branchless epilogue: C element r lives at (row = hf*8+r, col = m)
        #pragma unroll
        for (int r = 0; r < 8; ++r) {
            float x0 = c0[r] + si8[r][0] + sj[0];
            float x1 = c1[r] + si8[r][1] + sj[1];
            float x2 = c2[r] + si8[r][2] + sj[2];
            float x3 = c3[r] + si8[r][3] + sj[3];
            float e0 = fmaxf(x0, LEAKY * x0);   // leaky_relu as max(x, 0.2x)
            float e1 = fmaxf(x1, LEAKY * x1);
            float e2 = fmaxf(x2, LEAKY * x2);
            float e3 = fmaxf(x3, LEAKY * x3);
            int ad = adv[r];
            float val = MASKV;
            val = (ad == 1) ? e0 : val;
            val = (ad == 2) ? e1 : val;
            val = (ad == 3) ? e2 : val;
            val = (ad == 4) ? e3 : val;
            es[(hf * 8 + r) * N_DIM + j0 + m] = val;
        }
    }
    __syncthreads();

    // ---- softmax over j for each of the 16 rows (whole wave cooperates per row)
    for (int r = 0; r < 16; ++r) {
        v4f vv[4];
        #pragma unroll
        for (int q = 0; q < 4; ++q)
            vv[q] = *(const v4f*)&es[r * N_DIM + lane * 16 + q * 4];
        float mx = -3.0e38f;
        #pragma unroll
        for (int q = 0; q < 4; ++q)
            #pragma unroll
            for (int t = 0; t < 4; ++t) mx = fmaxf(mx, vv[q][t]);
        #pragma unroll
        for (int o = 16; o >= 1; o >>= 1) mx = fmaxf(mx, __shfl_xor(mx, o, 32));
        float sm = 0.f;
        #pragma unroll
        for (int q = 0; q < 4; ++q)
            #pragma unroll
            for (int t = 0; t < 4; ++t) {
                float ev = __expf(vv[q][t] - mx);
                vv[q][t] = ev;
                sm += ev;
            }
        #pragma unroll
        for (int o = 16; o >= 1; o >>= 1) sm += __shfl_xor(sm, o, 32);
        float inv = 1.0f / sm;
        v8bf o0, o1;
        #pragma unroll
        for (int t = 0; t < 8; ++t) {
            o0[t] = (__bf16)(vv[t >> 2][t & 3] * inv);
            o1[t] = (__bf16)(vv[2 + (t >> 2)][t & 3] * inv);
        }
        *(v8bf*)&alph[r * N_DIM + lane * 16]     = o0;
        *(v8bf*)&alph[r * N_DIM + lane * 16 + 8] = o1;
    }
    __syncthreads();

    // ---- out = alpha(16x512) @ h(512x128): K=512 in 16 steps, N=128 in 8 tiles
    v8f acc[8];
    v8f z = {0.f, 0.f, 0.f, 0.f, 0.f, 0.f, 0.f, 0.f};
    #pragma unroll
    for (int nt = 0; nt < 8; ++nt) acc[nt] = z;

    for (int kk = 0; kk < 16; ++kk) {
        int kb = kk * 32;
        v8bf lo = *(const v8bf*)&alph[m * N_DIM + kb + hf * 16];
        v8bf hi = *(const v8bf*)&alph[m * N_DIM + kb + hf * 16 + 8];
        v16bf af = __builtin_shufflevector(lo, hi, 0, 1, 2, 3, 4, 5, 6, 7,
                                                   8, 9, 10, 11, 12, 13, 14, 15);
        #pragma unroll
        for (int nt = 0; nt < 8; ++nt) {
            // B fragment from transposed h: lane = column d, elements = K (j) contiguous
            v16bf bf = *(const v16bf*)(hT + (((size_t)b * D_DIM) + nt * 16 + m) * N_DIM
                                          + kb + hf * 16);
            acc[nt] = __builtin_amdgcn_wmma_f32_16x16x32_bf16(false, af, false, bf, (short)0, acc[nt], false, false);
        }
    }

    #pragma unroll
    for (int nt = 0; nt < 8; ++nt) {
        #pragma unroll
        for (int r = 0; r < 8; ++r) {
            out[(((size_t)b * N_DIM) + i0 + hf * 8 + r) * D_DIM + nt * 16 + m] = acc[nt][r];
        }
    }
}

extern "C" void kernel_launch(void* const* d_in, const int* in_sizes, int n_in,
                              void* d_out, int out_size, void* d_ws, size_t ws_size,
                              hipStream_t stream) {
    const float* h_locs  = (const float*)d_in[0];
    const float* h_times = (const float*)d_in[1];
    const float* int_emb = (const float*)d_in[2];
    const float* h       = (const float*)d_in[3];
    const int*   adj     = (const int*)d_in[4];
    const float* wd      = (const float*)d_in[5];
    const float* a0      = (const float*)d_in[6];
    const float* a1      = (const float*)d_in[7];
    const float* a2      = (const float*)d_in[8];
    const float* a3      = (const float*)d_in[9];
    // d_in[10] = bias: present in the signature but unused by the reference.

    char* ws = (char*)d_ws;
    float*  hds  = (float*)ws;                               // 64*128*4      = 32 KB
    float*  sArr = (float*)(ws + 32768);                     // 64*512*4*4    = 512 KB
    __bf16* hB   = (__bf16*)(ws + 32768 + 524288);           // 64*512*128*2  = 8 MB
    __bf16* hT   = (__bf16*)(ws + 32768 + 524288 + 8388608); // 64*128*512*2  = 8 MB

    hipLaunchKernelGGL(k_hds,  dim3(B_DIM), dim3(D_DIM), 0, stream,
                       h_locs, h_times, int_emb, wd, hds);
    hipLaunchKernelGGL(k_prep, dim3((B_DIM * N_DIM) / 256), dim3(256), 0, stream,
                       h, hds, a0, a1, a2, a3, hB, hT, sArr);
    hipLaunchKernelGGL(k_attn, dim3(B_DIM * (N_DIM / 16)), dim3(32), 0, stream,
                       h, a0, a1, a2, a3, adj, hB, hT, sArr, (float*)d_out);
}
